// MPDecoder_83434034692199
// MI455X (gfx1250) — matrix operand order repaired and checked by reading, not compiled, not run on hardware
//
#include <hip/hip_runtime.h>

typedef __bf16 bf16_t;
typedef __bf16 v16bf __attribute__((ext_vector_type(16)));
typedef __bf16 v8bf  __attribute__((ext_vector_type(8)));
typedef float  v8f   __attribute__((ext_vector_type(8)));
typedef float  v4f   __attribute__((ext_vector_type(4)));

#define BATCH 32
#define NN    500
#define PP    500
#define HH    128
#define NPAD  512

__device__ __forceinline__ int imin(int a, int b) { return a < b ? a : b; }

__device__ __forceinline__ v8f bwmma(v16bf a, v16bf b, v8f c) {
  return __builtin_amdgcn_wmma_f32_16x16x32_bf16(false, a, false, b, (short)0, c, false, false);
}

// A-tile (16x32, M x K) per-lane load: lane holds row m, K = [kbase..kbase+15] (contiguous)
// lanerow points at this lane's row; off = 32*chunk + kbase
__device__ __forceinline__ v16bf load_a(const bf16_t* lanerow, int off) {
  v8bf lo = *(const v8bf*)(lanerow + off);
  v8bf hi = *(const v8bf*)(lanerow + off + 8);
  v16bf a;
#pragma unroll
  for (int i = 0; i < 8; ++i) { a[i] = lo[i]; a[8 + i] = hi[i]; }
  return a;
}

// ---------------------------------------------------------------------------
// Prep: bf16 copies of embeddings (row-major, padded+zeroed), E^T, q_first
// ---------------------------------------------------------------------------
__global__ void k_prep(const float* __restrict__ E, const float* __restrict__ QF,
                       bf16_t* __restrict__ Eb, bf16_t* __restrict__ Etr,
                       bf16_t* __restrict__ QFb) {
  int idx = blockIdx.x * blockDim.x + threadIdx.x;  // B*NPAD*HH threads
  int h = idx % HH;
  int row = (idx / HH) % NPAD;
  int b = idx / (HH * NPAD);
  if (b >= BATCH) return;
  float e = E[((size_t)b * NN + imin(row, NN - 1)) * HH + h];
  bf16_t eb = (bf16_t)(row < NN ? e : 0.0f);
  Eb[((size_t)b * NPAD + row) * HH + h] = eb;
  Etr[((size_t)b * HH + h) * NPAD + row] = eb;
  float q = QF[((size_t)b * PP + imin(row, PP - 1)) * HH + h];
  QFb[((size_t)b * NPAD + row) * HH + h] = (bf16_t)(row < PP ? q : 0.0f);
}

// mean of embeddings over N, per (b,h)
__global__ void k_meanE(const float* __restrict__ E, float* __restrict__ meanE) {
  int b = blockIdx.x, h = threadIdx.x;
  float s = 0.f;
  for (int n = 0; n < NN; ++n) s += E[((size_t)b * NN + n) * HH + h];
  meanE[b * HH + h] = s * (1.0f / NN);
}

// fused weights: Wa = Wq@Wq_last, Wc = (Wq@Wq_visited)/N  (+ bf16 copies of Wq, Wk)
__global__ void k_weights(const float* __restrict__ Wq, const float* __restrict__ Wql,
                          const float* __restrict__ Wqv, const float* __restrict__ Wk,
                          bf16_t* __restrict__ Wa, bf16_t* __restrict__ Wc,
                          bf16_t* __restrict__ Wqb, bf16_t* __restrict__ Wkb) {
  int h = blockIdx.x, j = threadIdx.x;
  float sa = 0.f, sc = 0.f;
  for (int t = 0; t < HH; ++t) {
    float wq = Wq[h * HH + t];
    sa += wq * Wql[t * HH + j];
    sc += wq * Wqv[t * HH + j];
  }
  Wa[h * HH + j] = (bf16_t)sa;
  Wc[h * HH + j] = (bf16_t)(sc * (1.0f / NN));
  Wqb[h * HH + j] = (bf16_t)Wq[h * HH + j];
  Wkb[h * HH + j] = (bf16_t)Wk[h * HH + j];
}

// per-batch graph-query bias: G[b,h] = ((meanE@Wqg^T)@Wq^T)[h]
__global__ void k_gterm(const float* __restrict__ meanE, const float* __restrict__ Wqg,
                        const float* __restrict__ Wq, float* __restrict__ G) {
  __shared__ float tg[HH];
  int b = blockIdx.x, t = threadIdx.x;
  float s = 0.f;
  for (int j = 0; j < HH; ++j) s += meanE[b * HH + j] * Wqg[t * HH + j];
  tg[t] = s;
  __syncthreads();
  float g = 0.f;
  for (int tt = 0; tt < HH; ++tt) g += tg[tt] * Wq[t * HH + tt];  // t reused as output h
  G[b * HH + t] = g;
}

// ---------------------------------------------------------------------------
// K = E @ Wk^T   -> Kb[b][n][h] bf16 (rows n>=NN are zero since Eb is padded)
// grid (B, NPAD/16), block 128 (4 waves, wave w does h-tiles {2w,2w+1})
// ---------------------------------------------------------------------------
__global__ void __launch_bounds__(128) k_kproj(const bf16_t* __restrict__ Eb,
                                               const bf16_t* __restrict__ Wkb,
                                               bf16_t* __restrict__ Kb) {
  int b = blockIdx.x, ntile = blockIdx.y;
  int tid = threadIdx.x, w = tid >> 5, lane = tid & 31;
  int mlo = lane & 15, hi = lane >> 4, kbase = 16 * hi;
  const bf16_t* Arow = Eb + ((size_t)b * NPAD + ntile * 16 + mlo) * HH;
  v16bf A[4];
#pragma unroll
  for (int c = 0; c < 4; ++c) A[c] = load_a(Arow, 32 * c + kbase);
#pragma unroll
  for (int ct = 0; ct < 2; ++ct) {
    int hcol = (2 * w + ct) * 16 + mlo;
    const bf16_t* Brow = Wkb + hcol * HH;  // B col = Wk row (contiguous)
    v8f acc = {};
#pragma unroll
    for (int c = 0; c < 4; ++c) {
      v16bf Bv = *(const v16bf*)(Brow + 32 * c + 16 * hi);
      acc = bwmma(A[c], Bv, acc);
    }
#pragma unroll
    for (int j = 0; j < 8; ++j) {
      int nrow = ntile * 16 + j + 8 * hi;
      Kb[((size_t)b * NPAD + nrow) * HH + hcol] = (bf16_t)acc[j];
    }
  }
}

// ---------------------------------------------------------------------------
// finalQ: phase1 U = visited @ E (reduce n), phase2 fused triple projection.
// grid (B, NPAD/16), block 128
// ---------------------------------------------------------------------------
__global__ void __launch_bounds__(128) k_final(const float* __restrict__ mask,
                                               const bf16_t* __restrict__ Etr,
                                               const bf16_t* __restrict__ Eb,
                                               const bf16_t* __restrict__ QFb,
                                               const bf16_t* __restrict__ Wa,
                                               const bf16_t* __restrict__ Wqb,
                                               const bf16_t* __restrict__ Wc,
                                               const int* __restrict__ last_node,
                                               const float* __restrict__ G,
                                               bf16_t* __restrict__ FQ) {
  __shared__ bf16_t Ut[16 * HH];
  int b = blockIdx.x, ptile = blockIdx.y;
  int tid = threadIdx.x, w = tid >> 5, lane = tid & 31;
  int mlo = lane & 15, hi = lane >> 4, kbase = 8 * hi;
  int p = ptile * 16 + mlo;
  const float* mrow = mask + ((size_t)b * PP + imin(p, PP - 1)) * NN;

  // ---- phase 1: U[16 x 128] = visited(16 x 512) @ E(512 x 128)
  int h0 = (2 * w) * 16 + mlo;
  const bf16_t* Bt0 = Etr + ((size_t)b * HH + h0) * NPAD;
  const bf16_t* Bt1 = Etr + ((size_t)b * HH + h0 + 16) * NPAD;
  v8f acc0 = {}, acc1 = {};
  for (int c = 0; c < 16; ++c) {
    int n0 = 32 * c;
    // A tile: convert mask (-inf -> 1, else 0) on the fly
    v16bf Av;
    v4f l0 = *(const v4f*)(mrow + n0 + kbase);          // lo half always valid (max idx 495)
    v4f l1 = *(const v4f*)(mrow + n0 + kbase + 4);
#pragma unroll
    for (int i = 0; i < 4; ++i) {
      Av[i] = (bf16_t)((l0[i] < -1e30f) ? 1.0f : 0.0f);
      Av[4 + i] = (bf16_t)((l1[i] < -1e30f) ? 1.0f : 0.0f);
    }
#pragma unroll
    for (int i = 0; i < 8; ++i) {  // hi half: clamp + select (last chunk crosses N)
      int k2 = n0 + 16 + kbase + i;
      float v = mrow[imin(k2, NN - 1)];
      Av[8 + i] = (bf16_t)((k2 < NN && v < -1e30f) ? 1.0f : 0.0f);
    }
    v16bf B0 = __builtin_shufflevector(*(const v8bf*)(Bt0 + n0 + kbase), *(const v8bf*)(Bt0 + n0 + 16 + kbase), 0, 1, 2, 3, 4, 5, 6, 7, 8, 9, 10, 11, 12, 13, 14, 15);
    v16bf B1 = __builtin_shufflevector(*(const v8bf*)(Bt1 + n0 + kbase), *(const v8bf*)(Bt1 + n0 + 16 + kbase), 0, 1, 2, 3, 4, 5, 6, 7, 8, 9, 10, 11, 12, 13, 14, 15);
    acc0 = bwmma(Av, B0, acc0);
    acc1 = bwmma(Av, B1, acc1);
  }
#pragma unroll
  for (int j = 0; j < 8; ++j) {
    int m = j + 8 * hi;
    Ut[m * HH + (2 * w) * 16 + mlo] = (bf16_t)acc0[j];
    Ut[m * HH + (2 * w + 1) * 16 + mlo] = (bf16_t)acc1[j];
  }
  __syncthreads();

  // ---- phase 2: final = U@Wc^T + LE@Wa^T + QF@Wq^T + G
  int ln = last_node[b * PP + imin(p, PP - 1)];
  const bf16_t* Urow = &Ut[mlo * HH];
  const bf16_t* LErow = Eb + ((size_t)b * NPAD + ln) * HH;
  const bf16_t* QFrow = QFb + ((size_t)b * NPAD + p) * HH;
#pragma unroll
  for (int ct = 0; ct < 2; ++ct) {
    int hcol = (2 * w + ct) * 16 + mlo;
    const bf16_t* BrowC = Wc + hcol * HH;
    const bf16_t* BrowA = Wa + hcol * HH;
    const bf16_t* BrowQ = Wqb + hcol * HH;
    v8f acc = {};
#pragma unroll
    for (int c = 0; c < 4; ++c) {
      int off = 32 * c + 16 * hi;
      int boff = 32 * c + 16 * hi;
      v16bf Au = load_a(Urow, off);
      v16bf Al = load_a(LErow, off);
      v16bf Aq = load_a(QFrow, off);
      acc = bwmma(Au, *(const v16bf*)(BrowC + boff), acc);
      acc = bwmma(Al, *(const v16bf*)(BrowA + boff), acc);
      acc = bwmma(Aq, *(const v16bf*)(BrowQ + boff), acc);
    }
    float g = G[b * HH + hcol];
#pragma unroll
    for (int j = 0; j < 8; ++j) {
      int prow = ptile * 16 + j + 8 * hi;
      FQ[((size_t)b * NPAD + prow) * HH + hcol] = (bf16_t)(acc[j] + g);
    }
  }
}

// ---------------------------------------------------------------------------
// scores + fused softmax: S = FQ @ K^T, epilogue tanh/mask/exp, row-normalize
// grid (B, NPAD/16), block 128; wave w handles n-tiles w, w+4, ..., w+28
// ---------------------------------------------------------------------------
__global__ void __launch_bounds__(128) k_score(const bf16_t* __restrict__ FQ,
                                               const bf16_t* __restrict__ Kb,
                                               const float* __restrict__ dists,
                                               const float* __restrict__ mask,
                                               const int* __restrict__ last_node,
                                               float* __restrict__ out) {
  __shared__ float rowsum[16];
  int b = blockIdx.x, ptile = blockIdx.y;
  int tid = threadIdx.x, w = tid >> 5, lane = tid & 31;
  int mlo = lane & 15, hi = lane >> 4, kbase = 16 * hi;
  if (tid < 16) rowsum[tid] = 0.f;
  __syncthreads();

  const float SCALE = 1.0f / (8.0f * 11.313708498984761f);  // 1/(8*sqrt(128))
  const float INVSQRT2 = 0.7071067811865476f;

  const bf16_t* Arow = FQ + ((size_t)b * NPAD + ptile * 16 + mlo) * HH;
  v16bf A[4];
#pragma unroll
  for (int c = 0; c < 4; ++c) A[c] = load_a(Arow, 32 * c + kbase);

  int pj[8];
  const float* drow[8];
#pragma unroll
  for (int j = 0; j < 8; ++j) {
    int p = ptile * 16 + j + 8 * hi;
    pj[j] = p;
    int ln = last_node[b * PP + imin(p, PP - 1)];
    drow[j] = dists + ((size_t)b * NN + ln) * NN;
  }

  float keep[8][8];
  float rpart[8] = {};
#pragma unroll
  for (int it = 0; it < 8; ++it) {
    int ntile = w + 4 * it;
    const bf16_t* Brow = Kb + ((size_t)b * NPAD + ntile * 16 + mlo) * HH;
    v8f acc = {};
#pragma unroll
    for (int c = 0; c < 4; ++c) {
      v16bf Bv = *(const v16bf*)(Brow + 32 * c + 16 * hi);
      acc = bwmma(A[c], Bv, acc);
    }
    int n = ntile * 16 + mlo;
    int nc = imin(n, NN - 1);
    bool nok = n < NN;
#pragma unroll
    for (int j = 0; j < 8; ++j) {
      float d = drow[j][nc];
      float mk = mask[((size_t)b * PP + imin(pj[j], PP - 1)) * NN + nc];
      float s = acc[j] * SCALE - d * INVSQRT2;
      float cval = 10.0f * tanhf(s);
      float u = nok ? __expf(cval + fmaxf(mk, -1e8f) - 10.0f) : 0.0f;
      keep[it][j] = u;
      rpart[j] += u;
    }
  }
  // reduce row sums across the 16 lanes sharing each row, then across waves
#pragma unroll
  for (int j = 0; j < 8; ++j) {
    float v = rpart[j];
    v += __shfl_xor(v, 1, 32);
    v += __shfl_xor(v, 2, 32);
    v += __shfl_xor(v, 4, 32);
    v += __shfl_xor(v, 8, 32);
    rpart[j] = v;
  }
  if (mlo == 0) {
#pragma unroll
    for (int j = 0; j < 8; ++j) atomicAdd(&rowsum[j + 8 * hi], rpart[j]);
  }
  __syncthreads();
  float inv[8];
#pragma unroll
  for (int j = 0; j < 8; ++j) inv[j] = 1.0f / fmaxf(rowsum[j + 8 * hi], 1e-30f);
#pragma unroll
  for (int it = 0; it < 8; ++it) {
    int n = (w + 4 * it) * 16 + mlo;
    if (n < NN) {
#pragma unroll
      for (int j = 0; j < 8; ++j)
        if (pj[j] < PP) out[((size_t)b * PP + pj[j]) * NN + n] = keep[it][j] * inv[j];
    }
  }
}

// ---------------------------------------------------------------------------
extern "C" void kernel_launch(void* const* d_in, const int* in_sizes, int n_in,
                              void* d_out, int out_size, void* d_ws, size_t ws_size,
                              hipStream_t stream) {
  (void)in_sizes; (void)n_in; (void)out_size; (void)ws_size;
  const float* emb = (const float*)d_in[0];
  const float* dists = (const float*)d_in[1];
  const float* qfirst = (const float*)d_in[2];
  const float* Wql = (const float*)d_in[3];
  const float* Wqv = (const float*)d_in[4];
  const float* Wqg = (const float*)d_in[5];
  const float* Wq = (const float*)d_in[6];
  const float* Wk = (const float*)d_in[7];
  const int* last_node = (const int*)d_in[8];
  const float* gmask = (const float*)d_in[9];
  float* out = (float*)d_out;

  char* ws = (char*)d_ws;
  size_t off = 0;
  auto take = [&](size_t bytes) -> char* {
    char* p = ws + off;
    off = (off + bytes + 255) & ~(size_t)255;
    return p;
  };
  bf16_t* Eb  = (bf16_t*)take((size_t)BATCH * NPAD * HH * 2);
  bf16_t* Etr = (bf16_t*)take((size_t)BATCH * HH * NPAD * 2);
  bf16_t* QFb = (bf16_t*)take((size_t)BATCH * NPAD * HH * 2);
  bf16_t* Kb  = (bf16_t*)take((size_t)BATCH * NPAD * HH * 2);
  bf16_t* FQ  = (bf16_t*)take((size_t)BATCH * NPAD * HH * 2);
  bf16_t* Wab = (bf16_t*)take((size_t)HH * HH * 2);
  bf16_t* Wcb = (bf16_t*)take((size_t)HH * HH * 2);
  bf16_t* Wqb = (bf16_t*)take((size_t)HH * HH * 2);
  bf16_t* Wkb = (bf16_t*)take((size_t)HH * HH * 2);
  float* meanE = (float*)take((size_t)BATCH * HH * 4);
  float* G = (float*)take((size_t)BATCH * HH * 4);

  k_prep<<<(BATCH * NPAD * HH) / 256, 256, 0, stream>>>(emb, qfirst, Eb, Etr, QFb);
  k_meanE<<<BATCH, HH, 0, stream>>>(emb, meanE);
  k_weights<<<HH, HH, 0, stream>>>(Wq, Wql, Wqv, Wk, Wab, Wcb, Wqb, Wkb);
  k_gterm<<<BATCH, HH, 0, stream>>>(meanE, Wqg, Wq, G);
  k_kproj<<<dim3(BATCH, NPAD / 16), 128, 0, stream>>>(Eb, Wkb, Kb);
  k_final<<<dim3(BATCH, NPAD / 16), 128, 0, stream>>>(gmask, Etr, Eb, QFb, Wab, Wqb, Wcb,
                                                      last_node, G, FQ);
  k_score<<<dim3(BATCH, NPAD / 16), 128, 0, stream>>>(FQ, Kb, dists, gmask, last_node, out);
}
